// _QuantLSTMLayer_37598143709983
// MI455X (gfx1250) — 8-bit matrix operand order repaired and checked by reading, compiled, not run on hardware
//
#include <hip/hip_runtime.h>
#include <cstdint>

#define B_  64
#define T_  512
#define I_  512
#define H_  512
#define WMAT_ELEMS (I_ * H_)
#define BH_ (B_ * H_)

typedef int v8i __attribute__((ext_vector_type(8)));

__device__ __forceinline__ float fq_(float x, float inv_s, float s) {
    float q = rintf(x * inv_s);
    q = fminf(fmaxf(q, -128.0f), 127.0f);
    return q * s;
}

__device__ __forceinline__ float sigmoid_(float x) {
    return 1.0f / (1.0f + expf(-x));
}

__device__ __forceinline__ v8i load_a16x64(const int8_t* __restrict__ base,
                                           int row0, size_t ld_bytes, int k0, int lane) {
    const int m  = lane & 15;
    const int hi = lane >> 4;
    v8i a;
#pragma unroll
    for (int v = 0; v < 8; ++v) {
        const int k = ((v >> 1) << 4) | ((v & 1) << 2) | (hi << 3);
        a[v] = *reinterpret_cast<const int*>(base + (size_t)(row0 + m) * ld_bytes + (size_t)(k0 + k));
    }
    return a;
}

__device__ __forceinline__ v8i load_b64x16_nt(const int8_t* __restrict__ w,
                                              int n0, size_t ld_bytes, int k0, int lane) {
    const int n  = lane & 15;
    const int hi = lane >> 4;
    v8i b;
#pragma unroll
    for (int v = 0; v < 8; ++v) {
        const int k = ((v >> 1) << 4) | ((v & 1) << 2) | (hi << 3);
        b[v] = __builtin_nontemporal_load(
            reinterpret_cast<const int*>(w + (size_t)(n0 + n) * ld_bytes + (size_t)(k0 + k)));
    }
    return b;
}

__global__ void quant_x_kernel(const float* __restrict__ x, int8_t* __restrict__ qx, int n) {
    for (int i = blockIdx.x * blockDim.x + threadIdx.x; i < n; i += gridDim.x * blockDim.x) {
        float q = rintf(x[i] * 128.0f);
        q = fminf(fmaxf(q, -128.0f), 127.0f);
        qx[i] = (int8_t)(int)q;
    }
}

__global__ void quant_w_kernel(const float* __restrict__ w0, const float* __restrict__ w1,
                               const float* __restrict__ w2, const float* __restrict__ w3,
                               const float* __restrict__ w4, const float* __restrict__ w5,
                               const float* __restrict__ w6, const float* __restrict__ w7,
                               int8_t* __restrict__ qw, int n) {
    const float* ws[8] = {w0, w1, w2, w3, w4, w5, w6, w7};
    for (int i = blockIdx.x * blockDim.x + threadIdx.x; i < n; i += gridDim.x * blockDim.x) {
        const int sel = i >> 18;
        const int off = i & (WMAT_ELEMS - 1);
        float q = rintf(ws[sel][off] * 128.0f);
        q = fminf(fmaxf(q, -128.0f), 127.0f);
        qw[i] = (int8_t)(int)q;
    }
}

__global__ void init_state_kernel(const float* __restrict__ h0, const float* __restrict__ c0,
                                  int8_t* __restrict__ qh, float* __restrict__ c, int n) {
    for (int i = blockIdx.x * blockDim.x + threadIdx.x; i < n; i += gridDim.x * blockDim.x) {
        float qh_f = rintf(h0[i] * 128.0f);
        qh_f = fminf(fmaxf(qh_f, -128.0f), 127.0f);
        qh[i] = (int8_t)(int)qh_f;
        c[i] = fq_(c0[i], 32.0f, 1.0f / 32.0f);
    }
}

__global__ void __launch_bounds__(128, 1)
xgemm_kernel(const int8_t* __restrict__ qx, const int8_t* __restrict__ qw,
             int* __restrict__ xg) {
    const int lane  = threadIdx.x & 31;
    const int wv    = threadIdx.x >> 5;
    const int wtile = blockIdx.x * 4 + wv;
    const int gate  = wtile & 3;
    const int ct    = (wtile >> 2) & 31;
    const int rt    = wtile >> 7;
    const int r0    = rt * 16;
    const int n0    = ct * 16;
    const int8_t* w = qw + (size_t)gate * WMAT_ELEMS;

    v8i acc = (v8i)(0);
#pragma unroll
    for (int kk = 0; kk < 8; ++kk) {
        const int k0 = kk * 64;
        const v8i a = load_a16x64(qx, r0, (size_t)I_, k0, lane);
        const v8i b = load_b64x16_nt(w, n0, (size_t)I_, k0, lane);
        acc = __builtin_amdgcn_wmma_i32_16x16x64_iu8(true, a, true, b, acc, false, false);
    }

    const int n  = n0 + (lane & 15);
    const int hi = lane >> 4;
#pragma unroll
    for (int r = 0; r < 8; ++r) {
        const int row = r0 + r + (hi << 3);
        const int b   = row >> 9;
        const int t   = row & (T_ - 1);
        __builtin_nontemporal_store(acc[r],
            xg + (((size_t)(t * 4 + gate)) << 15) + ((size_t)b << 9) + n);
    }
}

__global__ void __launch_bounds__(128, 1)
lstm_step_hoisted_kernel(const int* __restrict__ xg,
                         const int8_t* __restrict__ qwh,
                         const float* __restrict__ b_i, const float* __restrict__ b_f,
                         const float* __restrict__ b_c, const float* __restrict__ b_o,
                         const int8_t* __restrict__ qh_in,
                         int8_t* __restrict__ qh_out,
                         float* __restrict__ c,
                         float* __restrict__ ys,
                         float* __restrict__ hT, float* __restrict__ cT,
                         int t) {
    const int lane = threadIdx.x & 31;
    const int wv   = threadIdx.x >> 5;
    const int tile = blockIdx.x * 4 + wv;
    const int b0   = (tile >> 5) * 16;
    const int n0   = (tile & 31) * 16;
    const int nloc = lane & 15;
    const int hi   = lane >> 4;
    const int n    = n0 + nloc;

    v8i acc[4];
#pragma unroll
    for (int g = 0; g < 4; ++g) {
        const int* xgg = xg + (((size_t)(t * 4 + g)) << 15);
#pragma unroll
        for (int r = 0; r < 8; ++r)
            acc[g][r] = xgg[((size_t)(b0 + r + (hi << 3)) << 9) + n];
    }

    const int8_t* whh[4] = {qwh + 0 * (size_t)WMAT_ELEMS, qwh + 1 * (size_t)WMAT_ELEMS,
                            qwh + 2 * (size_t)WMAT_ELEMS, qwh + 3 * (size_t)WMAT_ELEMS};

#pragma unroll
    for (int kk = 0; kk < 8; ++kk) {
        const int k0 = kk * 64;
        const v8i ah = load_a16x64(qh_in, b0, (size_t)H_, k0, lane);
        v8i bh[4];
#pragma unroll
        for (int g = 0; g < 4; ++g) bh[g] = load_b64x16_nt(whh[g], n0, (size_t)H_, k0, lane);
#pragma unroll
        for (int g = 0; g < 4; ++g)
            acc[g] = __builtin_amdgcn_wmma_i32_16x16x64_iu8(true, ah, true, bh[g], acc[g], false, false);
    }

    const float S14  = 1.0f / 16384.0f;
    const float bi_n = b_i[n], bf_n = b_f[n], bc_n = b_c[n], bo_n = b_o[n];

#pragma unroll
    for (int r = 0; r < 8; ++r) {
        const int b = b0 + r + (hi << 3);
        const size_t bh_idx = (size_t)b * H_ + n;

        const float gi = (float)acc[0][r] * S14 + bi_n;
        const float gf = (float)acc[1][r] * S14 + bf_n;
        const float gc = (float)acc[2][r] * S14 + bc_n;
        const float go = (float)acc[3][r] * S14 + bo_n;

        const float iv = fq_(sigmoid_(fq_(gi, 16.0f, 0.0625f)), 128.0f, 1.0f / 128.0f);
        const float fv = fq_(sigmoid_(fq_(gf, 16.0f, 0.0625f)), 128.0f, 1.0f / 128.0f);
        const float gv = fq_(tanhf  (fq_(gc, 16.0f, 0.0625f)), 128.0f, 1.0f / 128.0f);
        const float ov = fq_(sigmoid_(fq_(go, 16.0f, 0.0625f)), 128.0f, 1.0f / 128.0f);

        const float c_old = c[bh_idx];
        const float fc    = fq_(fv * c_old, 32.0f, 1.0f / 32.0f);
        const float ic    = fq_(iv * gv,    32.0f, 1.0f / 32.0f);
        const float c_new = fq_(fc + ic,    32.0f, 1.0f / 32.0f);
        const float htanh = fq_(tanhf(c_new), 128.0f, 1.0f / 128.0f);
        const float h_new = fq_(ov * htanh, 128.0f, 1.0f / 128.0f);

        c[bh_idx] = c_new;
        ys[((size_t)b * T_ + t) * H_ + n] = h_new;
        qh_out[bh_idx] = (int8_t)(int)rintf(h_new * 128.0f);
        if (t == T_ - 1) { hT[bh_idx] = h_new; cT[bh_idx] = c_new; }
    }
}

__global__ void __launch_bounds__(128, 1)
lstm_step_fused_kernel(const int8_t* __restrict__ qx,
                       const int8_t* __restrict__ qw,
                       const float* __restrict__ b_i, const float* __restrict__ b_f,
                       const float* __restrict__ b_c, const float* __restrict__ b_o,
                       const int8_t* __restrict__ qh_in,
                       int8_t* __restrict__ qh_out,
                       float* __restrict__ c,
                       float* __restrict__ ys,
                       float* __restrict__ hT, float* __restrict__ cT,
                       int t) {
    const int lane = threadIdx.x & 31;
    const int wv   = threadIdx.x >> 5;
    const int tile = blockIdx.x * 4 + wv;
    const int b0   = (tile >> 5) * 16;
    const int n0   = (tile & 31) * 16;

    const int8_t* wih[4] = {qw + 0 * (size_t)WMAT_ELEMS, qw + 1 * (size_t)WMAT_ELEMS,
                            qw + 2 * (size_t)WMAT_ELEMS, qw + 3 * (size_t)WMAT_ELEMS};
    const int8_t* whh[4] = {qw + 4 * (size_t)WMAT_ELEMS, qw + 5 * (size_t)WMAT_ELEMS,
                            qw + 6 * (size_t)WMAT_ELEMS, qw + 7 * (size_t)WMAT_ELEMS};

    v8i acc[4];
#pragma unroll
    for (int g = 0; g < 4; ++g) acc[g] = (v8i)(0);

    const int8_t* xbase = qx + (size_t)t * I_;

#pragma unroll
    for (int kk = 0; kk < 8; ++kk) {
        const int k0 = kk * 64;
        const v8i ax = load_a16x64(xbase, b0, (size_t)T_ * I_, k0, lane);
        const v8i ah = load_a16x64(qh_in, b0, (size_t)H_,      k0, lane);
        v8i bx[4], bh[4];
#pragma unroll
        for (int g = 0; g < 4; ++g) bx[g] = load_b64x16_nt(wih[g], n0, (size_t)I_, k0, lane);
#pragma unroll
        for (int g = 0; g < 4; ++g) bh[g] = load_b64x16_nt(whh[g], n0, (size_t)H_, k0, lane);
#pragma unroll
        for (int g = 0; g < 4; ++g)
            acc[g] = __builtin_amdgcn_wmma_i32_16x16x64_iu8(true, ax, true, bx[g], acc[g], false, false);
#pragma unroll
        for (int g = 0; g < 4; ++g)
            acc[g] = __builtin_amdgcn_wmma_i32_16x16x64_iu8(true, ah, true, bh[g], acc[g], false, false);
    }

    const float S14  = 1.0f / 16384.0f;
    const int   n    = n0 + (lane & 15);
    const int   hi   = lane >> 4;
    const float bi_n = b_i[n], bf_n = b_f[n], bc_n = b_c[n], bo_n = b_o[n];

#pragma unroll
    for (int r = 0; r < 8; ++r) {
        const int b = b0 + r + (hi << 3);
        const size_t bh_idx = (size_t)b * H_ + n;

        const float gi = (float)acc[0][r] * S14 + bi_n;
        const float gf = (float)acc[1][r] * S14 + bf_n;
        const float gc = (float)acc[2][r] * S14 + bc_n;
        const float go = (float)acc[3][r] * S14 + bo_n;

        const float iv = fq_(sigmoid_(fq_(gi, 16.0f, 0.0625f)), 128.0f, 1.0f / 128.0f);
        const float fv = fq_(sigmoid_(fq_(gf, 16.0f, 0.0625f)), 128.0f, 1.0f / 128.0f);
        const float gv = fq_(tanhf  (fq_(gc, 16.0f, 0.0625f)), 128.0f, 1.0f / 128.0f);
        const float ov = fq_(sigmoid_(fq_(go, 16.0f, 0.0625f)), 128.0f, 1.0f / 128.0f);

        const float c_old = c[bh_idx];
        const float fc    = fq_(fv * c_old, 32.0f, 1.0f / 32.0f);
        const float ic    = fq_(iv * gv,    32.0f, 1.0f / 32.0f);
        const float c_new = fq_(fc + ic,    32.0f, 1.0f / 32.0f);
        const float htanh = fq_(tanhf(c_new), 128.0f, 1.0f / 128.0f);
        const float h_new = fq_(ov * htanh, 128.0f, 1.0f / 128.0f);

        c[bh_idx] = c_new;
        ys[((size_t)b * T_ + t) * H_ + n] = h_new;
        qh_out[bh_idx] = (int8_t)(int)rintf(h_new * 128.0f);
        if (t == T_ - 1) { hT[bh_idx] = h_new; cT[bh_idx] = c_new; }
    }
}

extern "C" void kernel_launch(void* const* d_in, const int* in_sizes, int n_in,
                              void* d_out, int out_size, void* d_ws, size_t ws_size,
                              hipStream_t stream) {
    const float* x    = (const float*)d_in[0];
    const float* h0   = (const float*)d_in[1];
    const float* c0   = (const float*)d_in[2];
    const float* w_ii = (const float*)d_in[3];
    const float* w_if = (const float*)d_in[4];
    const float* w_ic = (const float*)d_in[5];
    const float* w_io = (const float*)d_in[6];
    const float* w_hi = (const float*)d_in[7];
    const float* w_hf = (const float*)d_in[8];
    const float* w_hc = (const float*)d_in[9];
    const float* w_ho = (const float*)d_in[10];
    const float* b_i  = (const float*)d_in[11];
    const float* b_f  = (const float*)d_in[12];
    const float* b_c  = (const float*)d_in[13];
    const float* b_o  = (const float*)d_in[14];

    const size_t QX_BYTES = (size_t)B_ * T_ * I_;
    const size_t QW_BYTES = (size_t)8 * WMAT_ELEMS;
    const size_t QH_BYTES = (size_t)BH_;
    const size_t C_BYTES  = (size_t)BH_ * 4;
    const size_t BASE     = QX_BYTES + QW_BYTES + 2 * QH_BYTES + C_BYTES;
    const size_t XG_BYTES = (size_t)T_ * 4 * BH_ * 4;

    char* ws = (char*)d_ws;
    int8_t* qx  = (int8_t*)ws;
    int8_t* qw  = (int8_t*)(ws + QX_BYTES);
    int8_t* qh0 = (int8_t*)(ws + QX_BYTES + QW_BYTES);
    int8_t* qh1 = (int8_t*)(ws + QX_BYTES + QW_BYTES + QH_BYTES);
    float*  cbf = (float*) (ws + QX_BYTES + QW_BYTES + 2 * QH_BYTES);
    int*    xg  = (int*)   (ws + BASE);
    const int8_t* qwh = qw + 4 * (size_t)WMAT_ELEMS;

    float* ys = (float*)d_out;
    float* hT = ys + (size_t)B_ * T_ * H_;
    float* cT = hT + (size_t)BH_;

    {
        const int n = (int)QX_BYTES;
        quant_x_kernel<<<(n + 255) / 256, 256, 0, stream>>>(x, qx, n);
    }
    {
        const int n = (int)QW_BYTES;
        quant_w_kernel<<<(n + 255) / 256, 256, 0, stream>>>(
            w_ii, w_if, w_ic, w_io, w_hi, w_hf, w_hc, w_ho, qw, n);
    }
    {
        const int n = BH_;
        init_state_kernel<<<(n + 255) / 256, 256, 0, stream>>>(h0, c0, qh0, cbf, n);
    }

    const bool hoist = (ws_size >= BASE + XG_BYTES);

    if (hoist) {
        xgemm_kernel<<<65536, 128, 0, stream>>>(qx, qw, xg);
        for (int t = 0; t < T_; ++t) {
            const int8_t* hin  = (t & 1) ? qh1 : qh0;
            int8_t*       hout = (t & 1) ? qh0 : qh1;
            lstm_step_hoisted_kernel<<<32, 128, 0, stream>>>(
                xg, qwh, b_i, b_f, b_c, b_o, hin, hout, cbf, ys, hT, cT, t);
        }
    } else {
        for (int t = 0; t < T_; ++t) {
            const int8_t* hin  = (t & 1) ? qh1 : qh0;
            int8_t*       hout = (t & 1) ? qh0 : qh1;
            lstm_step_fused_kernel<<<32, 128, 0, stream>>>(
                qx, qw, b_i, b_f, b_c, b_o, hin, hout, cbf, ys, hT, cT, t);
        }
    }
}
